// FBGCN_Layer_83554293777022
// MI455X (gfx1250) — matrix operand order repaired and checked by reading, not compiled, not run on hardware
//
#include <hip/hip_runtime.h>

#define NN 8192
#define EE 262144
#define DD 256

typedef __attribute__((ext_vector_type(4)))  float    v4f;
typedef __attribute__((ext_vector_type(8)))  float    v8f;
typedef __attribute__((ext_vector_type(8)))  _Float16 v8h;
typedef __attribute__((ext_vector_type(16))) _Float16 v16h;

// ---------------------------------------------------------------------------
// B swizzle + f32->f16 convert into WMMA fragment order.
// B order used here for V_WMMA_F32_16X16X32_F16 (32x16 tile, wave32, contiguous k, same as A):
//   lane l, half h (0..15):  B[k0 + 16*(l>>4) + h][16*nt + (l&15)]
// Bsw half-index = ks*8448 + l*264 + nt*16 + h
//   (264 = 256 + 8 pad halves: per-lane stride 528B -> spreads LDS banks,
//    stays 16B aligned; per-ks stride 16896B)
// ---------------------------------------------------------------------------
__global__ void swizzle_b_kernel(const float* __restrict__ B,
                                 _Float16* __restrict__ S, int K)
{
    int i = blockIdx.x * blockDim.x + threadIdx.x;
    int total = (K >> 5) * 8448;
    if (i >= total) return;
    int ks = i / 8448;
    int r  = i - ks * 8448;
    int l  = r / 264;
    int q  = r - l * 264;
    _Float16 v = (_Float16)0.0f;
    if (q < 256) {
        int nt = q >> 4;
        int h  = q & 15;
        int k  = (ks << 5) + ((l >> 4) << 4) + h;
        int n  = (nt << 4) + (l & 15);
        v = (_Float16)B[(size_t)k * DD + n];
    }
    S[i] = v;
}

// ---------------------------------------------------------------------------
// C[M,256] = A_f32[M,K] @ B[K,256], B given pre-swizzled f16 (Bsw), fp32 acc.
// Block: 128 threads = 4 waves; each wave owns 2 strips of 16 rows
// (128 rows/block). K processed in 128-wide chunks staged to LDS (66 KB).
// grid = (M/128, nsplit); nsplit>1 -> atomicAdd into zeroed C (f32 adds
// commute but are not associative, so rounding can depend on add order).
// ---------------------------------------------------------------------------
#define CHUNK_U4 4224   // 4 ks * 8448 halves * 2B / 16B

__global__ void __launch_bounds__(128)
wmma_gemm_f16(const float* __restrict__ A, const _Float16* __restrict__ Bsw,
              float* __restrict__ C, int M, int K, int do_relu, int nsplit)
{
    extern __shared__ uint4 dynsmem4[];
    _Float16* sB = (_Float16*)dynsmem4;

    const int tid  = threadIdx.x;
    const int lane = tid & 31;
    const int wv   = tid >> 5;
    const int nl   = lane & 15;
    const int hl   = lane >> 4;            // 0 or 1

    const int klen    = K / nsplit;
    const int kbase   = blockIdx.y * klen;
    const int nchunks = klen >> 7;         // 128 K per chunk

    const int row0 = blockIdx.x * 128 + wv * 32;

    const float* __restrict__ Arow0 = A + (size_t)(row0 + nl) * K;
    const float* __restrict__ Arow1 = A + (size_t)(row0 + 16 + nl) * K;

    v8f acc[2][16] = {};

    const uint4* __restrict__ gsrc_base =
        (const uint4*)(Bsw + (size_t)(kbase >> 5) * 8448);

    for (int ch = 0; ch < nchunks; ++ch) {
        // ---- stage one 128-K chunk of fragment-ordered B (contiguous copy)
        const uint4* __restrict__ gsrc = gsrc_base + (size_t)ch * CHUNK_U4;
#pragma unroll
        for (int j = 0; j < 33; ++j)       // 33*128 = 4224
            dynsmem4[tid + 128 * j] = gsrc[tid + 128 * j];
        __syncthreads();

#pragma unroll
        for (int ksl = 0; ksl < 4; ++ksl) {
            const int k0 = kbase + (ch << 7) + (ksl << 5);
            // A fragments, f32->f16 on the fly.
            // lane half hl: halves 0..15 <- K [k0+16*hl .. +15]
            v16h a0, a1;
            {
                const float* p0 = Arow0 + k0 + (hl << 4);
                const float* p1 = Arow1 + k0 + (hl << 4);
                v8f f00 = *(const v8f*)(p0);
                v8f f01 = *(const v8f*)(p0 + 8);
                v8f f10 = *(const v8f*)(p1);
                v8f f11 = *(const v8f*)(p1 + 8);
#pragma unroll
                for (int h = 0; h < 8; ++h) {
                    a0[h]     = (_Float16)f00[h];
                    a0[8 + h] = (_Float16)f01[h];
                    a1[h]     = (_Float16)f10[h];
                    a1[8 + h] = (_Float16)f11[h];
                }
            }
            const _Float16* Bl = sB + ksl * 8448 + lane * 264;
#pragma unroll
            for (int nt = 0; nt < 16; ++nt) {
                v8h b0 = *(const v8h*)(Bl + (nt << 4));
                v8h b1 = *(const v8h*)(Bl + (nt << 4) + 8);
                v16h b = __builtin_shufflevector(b0, b1, 0, 1, 2, 3, 4, 5, 6, 7,
                                                 8, 9, 10, 11, 12, 13, 14, 15);
                acc[0][nt] = __builtin_amdgcn_wmma_f32_16x16x32_f16(
                    false, a0, false, b, (short)0, acc[0][nt], false, false);
                acc[1][nt] = __builtin_amdgcn_wmma_f32_16x16x32_f16(
                    false, a1, false, b, (short)0, acc[1][nt], false, false);
            }
        }
        __syncthreads();
    }

    // ---- epilogue: C/D layout: VGPR j -> row = base + 8*hl + j, col = 16*nt+nl
#pragma unroll
    for (int s = 0; s < 2; ++s) {
        const int rbase = row0 + s * 16 + (hl << 3);
#pragma unroll
        for (int nt = 0; nt < 16; ++nt) {
            const int col = (nt << 4) + nl;
#pragma unroll
            for (int j = 0; j < 8; ++j) {
                float v = acc[s][nt][j];
                float* dst = &C[(size_t)(rbase + j) * DD + col];
                if (nsplit > 1) {
                    atomicAdd(dst, v);
                } else {
                    if (do_relu) v = fmaxf(v, 0.0f);
                    *dst = v;
                }
            }
        }
    }
}

// ---------------------------------------------------------------------------
// GCN-branch helper kernels (fp32, memory-trivial)
// ---------------------------------------------------------------------------
__global__ void fill_kernel(float* __restrict__ p, float v, int n) {
    int i = blockIdx.x * blockDim.x + threadIdx.x;
    if (i < n) p[i] = v;
}

__global__ void deg_edges_kernel(const int* __restrict__ dst, float* __restrict__ deg, int e) {
    int i = blockIdx.x * blockDim.x + threadIdx.x;
    if (i < e) atomicAdd(&deg[dst[i]], 1.0f);
}

__global__ void dis_kernel(float* __restrict__ deg, int n) {
    int i = blockIdx.x * blockDim.x + threadIdx.x;
    if (i < n) {
        float d = deg[i];
        deg[i] = (d > 0.0f) ? rsqrtf(fmaxf(d, 1.0f)) : 0.0f;
    }
}

__global__ void scatter_kernel(const int* __restrict__ src, const int* __restrict__ dst,
                               const float* __restrict__ dis, const float* __restrict__ xw,
                               float* __restrict__ agg, int e)
{
    const int b = blockIdx.x;
    const int c = threadIdx.x;
    int s, d;
    if (b < e) { s = src[b]; d = dst[b]; }
    else       { s = b - e;  d = b - e;  }
    const float norm = dis[s] * dis[d];
    atomicAdd(&agg[(size_t)d * DD + c], xw[(size_t)s * DD + c] * norm);
}

__global__ void combine_kernel(const float* __restrict__ agg, const float* __restrict__ bconv,
                               const float* __restrict__ Hh,
                               const float* __restrict__ aL, const float* __restrict__ aH,
                               float* __restrict__ out, int total)
{
    int i = blockIdx.x * blockDim.x + threadIdx.x;
    if (i < total) {
        const int c = i & (DD - 1);
        const float hl = fmaxf(agg[i] + bconv[c], 0.0f);
        out[i] = aL[0] * hl + aH[0] * Hh[i];
    }
}

// ---------------------------------------------------------------------------
extern "C" void kernel_launch(void* const* d_in, const int* in_sizes, int n_in,
                              void* d_out, int out_size, void* d_ws, size_t ws_size,
                              hipStream_t stream)
{
    const float* x     = (const float*)d_in[0];
    const int*   eidx  = (const int*)d_in[1];   // [2, E]: row0 = src, row1 = dst
    const float* lap   = (const float*)d_in[2];
    const float* dinv  = (const float*)d_in[3];
    const float* Whigh = (const float*)d_in[4];
    const float* Wconv = (const float*)d_in[5];
    const float* bconv = (const float*)d_in[6];
    const float* aL    = (const float*)d_in[7];
    const float* aH    = (const float*)d_in[8];
    float* out = (float*)d_out;

    const size_t nd = (size_t)NN * DD;          // 2M floats per [N,256] buffer
    float* W0   = (float*)d_ws;                 // R, then T2
    float* W1   = W0 + nd;                      // T1, then Hh
    float* W2   = W1 + nd;                      // XW
    float* agg  = W2 + nd;                      // GCN accumulation
    float* deg  = agg + nd;                     // degree -> deg^{-1/2} (N floats)
    _Float16* SW = (_Float16*)(deg + NN);       // swizzled f16 B (<= 4.33 MB)

    const int* src = eidx;
    const int* dst = eidx + EE;

    const dim3 blk(128);
    const size_t smem = (size_t)CHUNK_U4 * 16;  // 67584 B

    auto swz_blocks = [](int K) { return ((K >> 5) * 8448 + 255) / 256; };

    // ---- High-pass: Hh = d_inv @ (lap @ (d_inv @ relu(x @ W_high))) ----
    swizzle_b_kernel<<<swz_blocks(DD), 256, 0, stream>>>(Whigh, SW, DD);
    wmma_gemm_f16<<<dim3(NN / 128, 1), blk, smem, stream>>>(x, SW, W0, NN, DD, 1, 1);   // R

    swizzle_b_kernel<<<swz_blocks(NN), 256, 0, stream>>>(W0, SW, NN);
    fill_kernel<<<(int)((nd + 255) / 256), 256, 0, stream>>>(W1, 0.0f, (int)nd);
    wmma_gemm_f16<<<dim3(NN / 128, 4), blk, smem, stream>>>(dinv, SW, W1, NN, NN, 0, 4); // T1

    swizzle_b_kernel<<<swz_blocks(NN), 256, 0, stream>>>(W1, SW, NN);
    fill_kernel<<<(int)((nd + 255) / 256), 256, 0, stream>>>(W0, 0.0f, (int)nd);
    wmma_gemm_f16<<<dim3(NN / 128, 4), blk, smem, stream>>>(lap, SW, W0, NN, NN, 0, 4);  // T2

    swizzle_b_kernel<<<swz_blocks(NN), 256, 0, stream>>>(W0, SW, NN);
    fill_kernel<<<(int)((nd + 255) / 256), 256, 0, stream>>>(W1, 0.0f, (int)nd);
    wmma_gemm_f16<<<dim3(NN / 128, 4), blk, smem, stream>>>(dinv, SW, W1, NN, NN, 0, 4); // Hh

    // ---- Low-pass: GCNConv ----
    swizzle_b_kernel<<<swz_blocks(DD), 256, 0, stream>>>(Wconv, SW, DD);
    wmma_gemm_f16<<<dim3(NN / 128, 1), blk, smem, stream>>>(x, SW, W2, NN, DD, 0, 1);    // XW

    fill_kernel<<<(int)((nd + 255) / 256), 256, 0, stream>>>(agg, 0.0f, (int)nd);
    fill_kernel<<<(NN + 255) / 256, 256, 0, stream>>>(deg, 1.0f, NN);   // self-loops
    deg_edges_kernel<<<(EE + 255) / 256, 256, 0, stream>>>(dst, deg, EE);
    dis_kernel<<<(NN + 255) / 256, 256, 0, stream>>>(deg, NN);
    scatter_kernel<<<EE + NN, DD, 0, stream>>>(src, dst, deg, W2, agg, EE);

    // ---- out = aL * relu(agg + b) + aH * Hh ----
    combine_kernel<<<(int)((nd + 255) / 256), 256, 0, stream>>>(agg, bconv, W1, aL, aH,
                                                                out, (int)nd);
}
